// BigramModel_30786325578003
// MI455X (gfx1250) — compile-verified, matrix operand order checked by reading, not run on hardware
//
#include <hip/hip_runtime.h>

// ---------------------------------------------------------------------------
// GPT-style transformer forward for MI455X (gfx1250), bf16 WMMA + fp32 accum.
// V=65 E=384 T=256 H=6 HS=64 L=3 B=128
// ---------------------------------------------------------------------------

typedef __attribute__((ext_vector_type(16))) __bf16 bf16x16;
typedef __attribute__((ext_vector_type(8)))  float  floatx8;

#define DEVI static __device__ __forceinline__

constexpr int Vv   = 65;
constexpr int Ee   = 384;
constexpr int Tt   = 256;
constexpr int Hh   = 6;
constexpr int HSs  = 64;
constexpr int Ll   = 3;
constexpr int Bb   = 128;
constexpr int E4   = 4 * Ee;     // 1536
constexpr int QKVN = 3 * Ee;     // 1152
constexpr int Mtok = Bb * Tt;    // 32768
constexpr int NPO  = 128;        // logits width padded 65 -> 128

union AFrag { bf16x16 v; unsigned u[8]; };

// Load a 16-bit WMMA operand fragment where the K dimension is contiguous in
// memory. p points at (row, k0). Matches ISA 16-bit A layout:
//   lanes 0-15 : K 0..7 in v0..3, K 16..23 in v4..7
//   lanes 16-31: K 8..15 in v0..3, K 24..31 in v4..7
DEVI bf16x16 load_frag_k(const __bf16* p, int kh /*lane>>4*/) {
  const unsigned* pu = (const unsigned*)p;
  AFrag f;
#pragma unroll
  for (int i = 0; i < 4; ++i) f.u[i]     = pu[kh * 4 + i];
#pragma unroll
  for (int i = 0; i < 4; ++i) f.u[4 + i] = pu[8 + kh * 4 + i];
  return f.v;
}

DEVI float wave_sum(float v) {
#pragma unroll
  for (int o = 16; o > 0; o >>= 1) v += __shfl_xor(v, o, 32);
  return v;
}
DEVI float wave_max(float v) {
#pragma unroll
  for (int o = 16; o > 0; o >>= 1) v = fmaxf(v, __shfl_xor(v, o, 32));
  return v;
}

// ---------------------------------------------------------------------------
// Elementwise / setup kernels
// ---------------------------------------------------------------------------

__global__ __launch_bounds__(256) void embed_kernel(
    const int* __restrict__ x, const float* __restrict__ tok,
    const float* __restrict__ pos, float* __restrict__ h) {
  size_t i = (size_t)blockIdx.x * 256 + threadIdx.x;
  if (i >= (size_t)Mtok * Ee) return;
  int e  = (int)(i % Ee);
  int bt = (int)(i / Ee);
  int t  = bt % Tt;
  h[i] = tok[(size_t)x[bt] * Ee + e] + pos[(size_t)t * Ee + e];
}

// Wq/Wk/Wv [L][H][E][HS] -> bf16, output-major [L][1152][384] (n-row, k-contig)
__global__ __launch_bounds__(256) void conv_qkv_kernel(
    const float* __restrict__ Wq, const float* __restrict__ Wk,
    const float* __restrict__ Wv, __bf16* __restrict__ dst) {
  size_t i = (size_t)blockIdx.x * 256 + threadIdx.x;
  if (i >= (size_t)Ll * QKVN * Ee) return;
  int e  = (int)(i % Ee);
  int j  = (int)((i / Ee) % QKVN);
  int l  = (int)(i / ((size_t)Ee * QKVN));
  int sect = j / Ee;
  int jj   = j % Ee;
  int hh   = jj / HSs;
  int d    = jj % HSs;
  const float* src = (sect == 0) ? Wq : (sect == 1) ? Wk : Wv;
  dst[i] = (__bf16)src[(((size_t)l * Hh + hh) * Ee + e) * HSs + d];
}

// src [Lb][R][C] fp32 -> dst [Lb][C][R] bf16  (transpose + convert)
__global__ __launch_bounds__(256) void conv_t_kernel(
    const float* __restrict__ src, __bf16* __restrict__ dst,
    int R, int C, int Lb) {
  size_t total = (size_t)Lb * R * C;
  size_t i = (size_t)blockIdx.x * 256 + threadIdx.x;
  if (i >= total) return;
  int r = (int)(i % R);
  int c = (int)((i / R) % C);
  int l = (int)(i / ((size_t)R * C));
  dst[i] = (__bf16)src[((size_t)l * R + r) * C + c];
}

// Wout [E][65] -> bf16 [128][384] (output-major, zero-padded rows 65..127)
__global__ __launch_bounds__(256) void conv_wout_kernel(
    const float* __restrict__ Wout, __bf16* __restrict__ dst) {
  size_t i = (size_t)blockIdx.x * 256 + threadIdx.x;
  if (i >= (size_t)NPO * Ee) return;
  int e = (int)(i % Ee);
  int n = (int)(i / Ee);
  dst[i] = (n < Vv) ? (__bf16)Wout[(size_t)e * Vv + n] : (__bf16)0.0f;
}

// One wave per row: LayerNorm fp32 -> bf16
__global__ __launch_bounds__(256) void ln_kernel(
    const float* __restrict__ X, const float* __restrict__ g,
    const float* __restrict__ bta, __bf16* __restrict__ Y) {
  size_t row = (size_t)blockIdx.x * 8 + (threadIdx.x >> 5);
  int lane = threadIdx.x & 31;
  const float* xp = X + row * Ee;
  float v[12];
#pragma unroll
  for (int i = 0; i < 12; ++i) v[i] = xp[lane + 32 * i];
  float s = 0.f;
#pragma unroll
  for (int i = 0; i < 12; ++i) s += v[i];
  float mean = wave_sum(s) * (1.0f / Ee);
  float ss = 0.f;
#pragma unroll
  for (int i = 0; i < 12; ++i) { float d = v[i] - mean; ss += d * d; }
  float inv = rsqrtf(wave_sum(ss) * (1.0f / Ee) + 1e-5f);
  __bf16* yp = Y + row * Ee;
#pragma unroll
  for (int i = 0; i < 12; ++i) {
    int c = lane + 32 * i;
    yp[c] = (__bf16)((v[i] - mean) * inv * g[c] + bta[c]);
  }
}

// ---------------------------------------------------------------------------
// Generic bf16 WMMA GEMM:  C[M x N] = A[M x K] * Bt[N x K]^T (+bias,+resid,relu)
// Block = 8 waves, wave tile 32x64 (2x4 WMMA tiles), block tile 128x128.
// ---------------------------------------------------------------------------
template <bool HAS_BIAS, bool RELU, bool RESID, bool STORE_BF16>
__global__ __launch_bounds__(256) void gemm_kernel(
    const __bf16* __restrict__ A,   // [M x K] row-major
    const __bf16* __restrict__ Bt,  // [Npad x K] row-major (B transposed)
    const float* __restrict__ bias, // [>= Nstore] or null
    float* __restrict__ Cf,         // fp32 out (and resid in) [M x ldc]
    __bf16* __restrict__ Cb,        // bf16 out [M x ldc]
    int M, int Npad, int K, int Nstore, int ldc) {
  const int lane = threadIdx.x & 31;
  const int wave = threadIdx.x >> 5;
  const int wm = wave & 3, wn = wave >> 2;
  const int nBN = Npad >> 7;
  const int bm = (int)blockIdx.x / nBN, bn = (int)blockIdx.x % nBN;
  const int row0 = bm * 128 + wm * 32;
  const int col0 = bn * 128 + wn * 64;
  const int kh = lane >> 4, l16 = lane & 15;

  floatx8 acc[2][4];
#pragma unroll
  for (int mt = 0; mt < 2; ++mt)
#pragma unroll
    for (int nt = 0; nt < 4; ++nt) acc[mt][nt] = (floatx8)0.0f;

  const __bf16* Arow[2];
#pragma unroll
  for (int mt = 0; mt < 2; ++mt)
    Arow[mt] = A + (size_t)(row0 + mt * 16 + l16) * K;
  const __bf16* Brow[4];
#pragma unroll
  for (int nt = 0; nt < 4; ++nt)
    Brow[nt] = Bt + (size_t)(col0 + nt * 16 + l16) * K;

  for (int k0 = 0; k0 < K; k0 += 32) {
    bf16x16 af[2], bf[4];
#pragma unroll
    for (int mt = 0; mt < 2; ++mt) af[mt] = load_frag_k(Arow[mt] + k0, kh);
#pragma unroll
    for (int nt = 0; nt < 4; ++nt) bf[nt] = load_frag_k(Brow[nt] + k0, kh);
#pragma unroll
    for (int mt = 0; mt < 2; ++mt)
#pragma unroll
      for (int nt = 0; nt < 4; ++nt)
        acc[mt][nt] = __builtin_amdgcn_wmma_f32_16x16x32_bf16(
            false, af[mt], false, bf[nt], (short)0, acc[mt][nt], false, false);
  }

#pragma unroll
  for (int nt = 0; nt < 4; ++nt) {
    int n = col0 + nt * 16 + l16;
    if (n >= Nstore) continue;
    float bv = HAS_BIAS ? bias[n] : 0.0f;
#pragma unroll
    for (int mt = 0; mt < 2; ++mt) {
      int mbase = row0 + mt * 16 + kh * 8;
#pragma unroll
      for (int r = 0; r < 8; ++r) {
        int m = mbase + r;
        float vv = acc[mt][nt][r] + bv;
        if (RESID) vv += Cf[(size_t)m * ldc + n];
        if (RELU) vv = vv > 0.0f ? vv : 0.0f;
        if (STORE_BF16) Cb[(size_t)m * ldc + n] = (__bf16)vv;
        else            Cf[(size_t)m * ldc + n] = vv;
      }
    }
  }
}

// ---------------------------------------------------------------------------
// Attention
// ---------------------------------------------------------------------------

// S[bh, t, s] = (q_t . k_s) * HS^-0.5, causal-masked. grid (4, B*H).
__global__ __launch_bounds__(256) void attn_score_kernel(
    const __bf16* __restrict__ qkv, float* __restrict__ S) {
  const int lane = threadIdx.x & 31;
  const int wave = threadIdx.x >> 5;
  const int wm = wave & 3, wn = wave >> 2;
  const int bh = blockIdx.y;
  const int bq = (int)blockIdx.x >> 1, bk = (int)blockIdx.x & 1;
  const int b = bh / Hh, hh = bh % Hh;
  const int row0 = bq * 128 + wm * 32;   // query index t
  const int col0 = bk * 128 + wn * 64;   // key index s
  const int kh = lane >> 4, l16 = lane & 15;

  floatx8 acc[2][4];
#pragma unroll
  for (int mt = 0; mt < 2; ++mt)
#pragma unroll
    for (int nt = 0; nt < 4; ++nt) acc[mt][nt] = (floatx8)0.0f;

  const __bf16* Ap[2];
#pragma unroll
  for (int mt = 0; mt < 2; ++mt)
    Ap[mt] = qkv + (size_t)(b * Tt + row0 + mt * 16 + l16) * QKVN + hh * HSs;
  const __bf16* Bp[4];
#pragma unroll
  for (int nt = 0; nt < 4; ++nt)
    Bp[nt] = qkv + (size_t)(b * Tt + col0 + nt * 16 + l16) * QKVN + Ee + hh * HSs;

#pragma unroll
  for (int k0 = 0; k0 < HSs; k0 += 32) {
    bf16x16 af[2], bf[4];
#pragma unroll
    for (int mt = 0; mt < 2; ++mt) af[mt] = load_frag_k(Ap[mt] + k0, kh);
#pragma unroll
    for (int nt = 0; nt < 4; ++nt) bf[nt] = load_frag_k(Bp[nt] + k0, kh);
#pragma unroll
    for (int mt = 0; mt < 2; ++mt)
#pragma unroll
      for (int nt = 0; nt < 4; ++nt)
        acc[mt][nt] = __builtin_amdgcn_wmma_f32_16x16x32_bf16(
            false, af[mt], false, bf[nt], (short)0, acc[mt][nt], false, false);
  }

  const float scale = 0.125f;  // 64^-0.5
#pragma unroll
  for (int nt = 0; nt < 4; ++nt) {
    int s = col0 + nt * 16 + l16;
#pragma unroll
    for (int mt = 0; mt < 2; ++mt) {
#pragma unroll
      for (int r = 0; r < 8; ++r) {
        int t = row0 + mt * 16 + kh * 8 + r;
        float v = (s <= t) ? acc[mt][nt][r] * scale : -1e30f;
        S[((size_t)bh * Tt + t) * Tt + s] = v;
      }
    }
  }
}

// One wave per row: softmax over 256 scores -> bf16 probs.
__global__ __launch_bounds__(256) void softmax_kernel(
    const float* __restrict__ S, __bf16* __restrict__ P) {
  size_t row = (size_t)blockIdx.x * 8 + (threadIdx.x >> 5);
  int lane = threadIdx.x & 31;
  const float* sp = S + row * Tt;
  float x[8];
#pragma unroll
  for (int i = 0; i < 8; ++i) x[i] = sp[lane + 32 * i];
  float mx = x[0];
#pragma unroll
  for (int i = 1; i < 8; ++i) mx = fmaxf(mx, x[i]);
  mx = wave_max(mx);
  float sum = 0.f;
#pragma unroll
  for (int i = 0; i < 8; ++i) { x[i] = __expf(x[i] - mx); sum += x[i]; }
  float inv = 1.0f / wave_sum(sum);
  __bf16* pp = P + row * Tt;
#pragma unroll
  for (int i = 0; i < 8; ++i) pp[lane + 32 * i] = (__bf16)(x[i] * inv);
}

// Vt[bh][d][s] = V[bh][s][d]  (so P*V GEMM can do contiguous-K loads)
__global__ __launch_bounds__(256) void vtrans_kernel(
    const __bf16* __restrict__ qkv, __bf16* __restrict__ vt) {
  size_t total = (size_t)Bb * Hh * HSs * Tt;
  size_t i = (size_t)blockIdx.x * 256 + threadIdx.x;
  if (i >= total) return;
  int s = (int)(i % Tt);
  size_t r = i / Tt;
  int d = (int)(r % HSs);
  int bh = (int)(r / HSs);
  int b = bh / Hh, hh = bh % Hh;
  vt[i] = qkv[(size_t)(b * Tt + s) * QKVN + 2 * Ee + hh * HSs + d];
}

// O[bh] = P[bh] (256x256) * V[bh] (256x64); write head-concat layout. grid B*H.
__global__ __launch_bounds__(256) void attn_out_kernel(
    const __bf16* __restrict__ P, const __bf16* __restrict__ Vt,
    __bf16* __restrict__ att) {
  const int lane = threadIdx.x & 31;
  const int wave = threadIdx.x >> 5;
  const int bh = blockIdx.x;
  const int b = bh / Hh, hh = bh % Hh;
  const int row0 = wave * 32;
  const int kh = lane >> 4, l16 = lane & 15;

  floatx8 acc[2][4];
#pragma unroll
  for (int mt = 0; mt < 2; ++mt)
#pragma unroll
    for (int nt = 0; nt < 4; ++nt) acc[mt][nt] = (floatx8)0.0f;

  const __bf16* Ap[2];
#pragma unroll
  for (int mt = 0; mt < 2; ++mt)
    Ap[mt] = P + (size_t)bh * Tt * Tt + (size_t)(row0 + mt * 16 + l16) * Tt;
  const __bf16* Bp[4];
#pragma unroll
  for (int nt = 0; nt < 4; ++nt)
    Bp[nt] = Vt + (size_t)bh * HSs * Tt + (size_t)(nt * 16 + l16) * Tt;

  for (int k0 = 0; k0 < Tt; k0 += 32) {
    bf16x16 af[2], bf[4];
#pragma unroll
    for (int mt = 0; mt < 2; ++mt) af[mt] = load_frag_k(Ap[mt] + k0, kh);
#pragma unroll
    for (int nt = 0; nt < 4; ++nt) bf[nt] = load_frag_k(Bp[nt] + k0, kh);
#pragma unroll
    for (int mt = 0; mt < 2; ++mt)
#pragma unroll
      for (int nt = 0; nt < 4; ++nt)
        acc[mt][nt] = __builtin_amdgcn_wmma_f32_16x16x32_bf16(
            false, af[mt], false, bf[nt], (short)0, acc[mt][nt], false, false);
  }

#pragma unroll
  for (int nt = 0; nt < 4; ++nt) {
    int d = nt * 16 + l16;
#pragma unroll
    for (int mt = 0; mt < 2; ++mt) {
#pragma unroll
      for (int r = 0; r < 8; ++r) {
        int t = row0 + mt * 16 + kh * 8 + r;
        att[(size_t)(b * Tt + t) * Ee + hh * HSs + d] = (__bf16)acc[mt][nt][r];
      }
    }
  }
}

// ---------------------------------------------------------------------------
// Host orchestration
// ---------------------------------------------------------------------------
extern "C" void kernel_launch(void* const* d_in, const int* in_sizes, int n_in,
                              void* d_out, int out_size, void* d_ws,
                              size_t ws_size, hipStream_t stream) {
  const int*   x       = (const int*)d_in[0];
  const float* tok_emb = (const float*)d_in[1];
  const float* pos_emb = (const float*)d_in[2];
  const float* Wq      = (const float*)d_in[3];
  const float* Wk      = (const float*)d_in[4];
  const float* Wv      = (const float*)d_in[5];
  const float* Wproj   = (const float*)d_in[6];
  const float* bproj   = (const float*)d_in[7];
  const float* ln1_g   = (const float*)d_in[8];
  const float* ln1_b   = (const float*)d_in[9];
  const float* ln2_g   = (const float*)d_in[10];
  const float* ln2_b   = (const float*)d_in[11];
  const float* W1      = (const float*)d_in[12];
  const float* b1      = (const float*)d_in[13];
  const float* W2      = (const float*)d_in[14];
  const float* b2      = (const float*)d_in[15];
  const float* lnf_g   = (const float*)d_in[16];
  const float* lnf_b   = (const float*)d_in[17];
  const float* Wout    = (const float*)d_in[18];
  const float* bout    = (const float*)d_in[19];
  float* out = (float*)d_out;
  (void)in_sizes; (void)n_in; (void)out_size; (void)ws_size;

  char* ws = (char*)d_ws;
  size_t off = 0;
  auto carve = [&](size_t bytes) -> char* {
    off = (off + 255) & ~(size_t)255;
    char* p = ws + off;
    off += bytes;
    return p;
  };
  float*  h     = (float*)carve((size_t)Mtok * Ee * 4);
  __bf16* xn    = (__bf16*)carve((size_t)Mtok * Ee * 2);
  __bf16* qkv   = (__bf16*)carve((size_t)Mtok * QKVN * 2);
  float*  sc    = (float*)carve((size_t)Bb * Hh * Tt * Tt * 4);
  __bf16* pr    = (__bf16*)carve((size_t)Bb * Hh * Tt * Tt * 2);
  __bf16* vt    = (__bf16*)carve((size_t)Bb * Hh * HSs * Tt * 2);
  __bf16* att   = (__bf16*)carve((size_t)Mtok * Ee * 2);
  __bf16* mid   = (__bf16*)carve((size_t)Mtok * E4 * 2);
  __bf16* wqkvT = (__bf16*)carve((size_t)Ll * QKVN * Ee * 2);
  __bf16* wprjT = (__bf16*)carve((size_t)Ll * Ee * Ee * 2);
  __bf16* w1T   = (__bf16*)carve((size_t)Ll * E4 * Ee * 2);
  __bf16* w2T   = (__bf16*)carve((size_t)Ll * Ee * E4 * 2);
  __bf16* woutT = (__bf16*)carve((size_t)NPO * Ee * 2);

  auto g1 = [](size_t n) { return dim3((unsigned)((n + 255) / 256)); };

  embed_kernel<<<g1((size_t)Mtok * Ee), 256, 0, stream>>>(x, tok_emb, pos_emb, h);
  conv_qkv_kernel<<<g1((size_t)Ll * QKVN * Ee), 256, 0, stream>>>(Wq, Wk, Wv, wqkvT);
  conv_t_kernel<<<g1((size_t)Ll * Ee * Ee), 256, 0, stream>>>(Wproj, wprjT, Ee, Ee, Ll);
  conv_t_kernel<<<g1((size_t)Ll * Ee * E4), 256, 0, stream>>>(W1, w1T, Ee, E4, Ll);
  conv_t_kernel<<<g1((size_t)Ll * E4 * Ee), 256, 0, stream>>>(W2, w2T, E4, Ee, Ll);
  conv_wout_kernel<<<g1((size_t)NPO * Ee), 256, 0, stream>>>(Wout, woutT);

  for (int l = 0; l < Ll; ++l) {
    ln_kernel<<<Mtok / 8, 256, 0, stream>>>(h, ln1_g + l * Ee, ln1_b + l * Ee, xn);
    gemm_kernel<false, false, false, true>
        <<<dim3((Mtok / 128) * (QKVN / 128)), 256, 0, stream>>>(
            xn, wqkvT + (size_t)l * QKVN * Ee, nullptr, nullptr, qkv,
            Mtok, QKVN, Ee, QKVN, QKVN);
    attn_score_kernel<<<dim3(4, Bb * Hh), 256, 0, stream>>>(qkv, sc);
    softmax_kernel<<<(Bb * Hh * Tt) / 8, 256, 0, stream>>>(sc, pr);
    vtrans_kernel<<<g1((size_t)Bb * Hh * HSs * Tt), 256, 0, stream>>>(qkv, vt);
    attn_out_kernel<<<Bb * Hh, 256, 0, stream>>>(pr, vt, att);
    gemm_kernel<true, false, true, false>
        <<<dim3((Mtok / 128) * (Ee / 128)), 256, 0, stream>>>(
            att, wprjT + (size_t)l * Ee * Ee, bproj + l * Ee, h, nullptr,
            Mtok, Ee, Ee, Ee, Ee);
    ln_kernel<<<Mtok / 8, 256, 0, stream>>>(h, ln2_g + l * Ee, ln2_b + l * Ee, xn);
    gemm_kernel<true, true, false, true>
        <<<dim3((Mtok / 128) * (E4 / 128)), 256, 0, stream>>>(
            xn, w1T + (size_t)l * E4 * Ee, b1 + l * E4, nullptr, mid,
            Mtok, E4, Ee, E4, E4);
    gemm_kernel<true, false, true, false>
        <<<dim3((Mtok / 128) * (Ee / 128)), 256, 0, stream>>>(
            mid, w2T + (size_t)l * Ee * E4, b2 + l * Ee, h, nullptr,
            Mtok, Ee, E4, Ee, Ee);
  }

  ln_kernel<<<Mtok / 8, 256, 0, stream>>>(h, lnf_g, lnf_b, xn);
  gemm_kernel<true, false, false, false>
      <<<dim3((Mtok / 128) * (NPO / 128)), 256, 0, stream>>>(
          xn, woutT, bout, out, nullptr, Mtok, NPO, Ee, Vv, Vv);
}
